// ALiBiMultiHeadAttention_18219251269911
// MI455X (gfx1250) — matrix operand order repaired and checked by reading, not compiled, not run on hardware
//
#include <hip/hip_runtime.h>
#include <hip/hip_bf16.h>

typedef __attribute__((ext_vector_type(8)))  __bf16 v8bf;
typedef __attribute__((ext_vector_type(16))) __bf16 v16bf;
typedef __attribute__((ext_vector_type(8)))  float  v8f;
typedef __attribute__((vector_size(16)))     int    v4i;

#define S_LEN 2048
#define EMBD  1024
#define NHEAD 16
#define HDIM  64
#define BSZ   2
#define LOG2E 1.4426950408889634f

#define AS1 __attribute__((address_space(1)))
#define AS3 __attribute__((address_space(3)))

#if defined(__has_builtin)
#if __has_builtin(__builtin_amdgcn_global_load_async_to_lds_b128)
#define USE_ASYNC_LDS 1
#endif
#endif

__device__ __forceinline__ unsigned short f2bf(float f) {
    unsigned int u = __float_as_uint(f);
    u += 0x7FFFu + ((u >> 16) & 1u);   // round to nearest even
    return (unsigned short)(u >> 16);
}

__device__ __forceinline__ v8bf ldv8(const unsigned short* p) {
    return *reinterpret_cast<const v8bf*>(p);
}

__device__ __forceinline__ v16bf comb(v8bf a, v8bf b) {
    return __builtin_shufflevector(a, b, 0,1,2,3,4,5,6,7,8,9,10,11,12,13,14,15);
}

// A operand (16x32, lane = row M): elems 0..7 = K[kbase+half*8..], 8..15 = K[kbase+16+half*8..]
__device__ __forceinline__ v16bf load_a(const unsigned short* row, int kbase, int half) {
    return comb(ldv8(row + kbase + half * 8), ldv8(row + kbase + 16 + half * 8));
}
// B operand (32x16, lane = col N): elems 0..7 = K[kbase+half*8..], 8..15 = K[kbase+16+half*8..]
__device__ __forceinline__ v16bf load_b(const unsigned short* row, int kbase, int half) {
    return comb(ldv8(row + kbase + half * 8), ldv8(row + kbase + 16 + half * 8));
}

__device__ __forceinline__ v8f wmma_bf16(v16bf a, v16bf b, v8f c) {
    return __builtin_amdgcn_wmma_f32_16x16x32_bf16(false, a, false, b, (short)0, c, false, false);
}

// ---------------- fp32 -> bf16 conversion ----------------
__global__ void cvt_bf16_kernel(const float* __restrict__ in, unsigned short* __restrict__ out, int n) {
    int i = blockIdx.x * 256 + threadIdx.x;
    if (i < n) out[i] = f2bf(in[i]);
}

// ---------------- QKV GEMM: [4096,3072] = Xbf[4096,1024] * Wbf[3072,1024]^T ----------------
// Block = 256 rows x 64 cols (8 waves x 32 rows). W tile (64x32) staged in LDS per k-step
// via async global->LDS copy. K/Q thirds stored to kqv row-major; V third stored directly
// transposed to vt[b,h,d,s].
__global__ __launch_bounds__(256) void qkv_gemm_kernel(const unsigned short* __restrict__ xbf,
                                                       unsigned short* __restrict__ wbf,
                                                       unsigned short* __restrict__ kqv,
                                                       unsigned short* __restrict__ vt) {
    const int tid  = threadIdx.x;
    const int lane = tid & 31;
    const int half = lane >> 4;
    const int l15  = lane & 15;
    const int wave = tid >> 5;
    const int bm   = blockIdx.x & 15;   // 16 M-blocks of 256 rows
    const int nt   = blockIdx.x >> 4;   // 48 N-tiles of 64 cols
    const int m0   = bm * 256 + wave * 32;
    const int n0   = nt * 64;

    __shared__ __align__(16) unsigned short bstage[64 * 32];

    const unsigned short* arow0 = xbf + (size_t)(m0 + l15) * EMBD;
    const unsigned short* arow1 = xbf + (size_t)(m0 + 16 + l15) * EMBD;

    v8f acc[2][4];
#pragma unroll
    for (int r = 0; r < 2; ++r)
#pragma unroll
        for (int t = 0; t < 4; ++t)
#pragma unroll
            for (int i = 0; i < 8; ++i) acc[r][t][i] = 0.0f;

    // this thread's 16B chunk of the W tile: row tid>>2 of 64, k-chunk (tid&3)*8
    unsigned short* gsrc = wbf + (size_t)(n0 + (tid >> 2)) * EMBD + (tid & 3) * 8;
    unsigned short* ldst = bstage + (tid >> 2) * 32 + (tid & 3) * 8;

    for (int k0 = 0; k0 < EMBD; k0 += 32) {
        __syncthreads();   // WAR: previous iteration's reads done
#if USE_ASYNC_LDS
        __builtin_amdgcn_global_load_async_to_lds_b128((AS1 v4i*)(gsrc + k0), (AS3 v4i*)ldst, 0, 0);
#if __has_builtin(__builtin_amdgcn_s_wait_asynccnt)
        __builtin_amdgcn_s_wait_asynccnt(0);
#else
        asm volatile("s_wait_asynccnt 0" ::: "memory");
#endif
#else
        *reinterpret_cast<v8bf*>(ldst) = *reinterpret_cast<const v8bf*>(gsrc + k0);
#endif
        __syncthreads();   // RAW: tile staged

        v16bf a0 = load_a(arow0, k0, half);
        v16bf a1 = load_a(arow1, k0, half);
#pragma unroll
        for (int t = 0; t < 4; ++t) {
            const unsigned short* bl = bstage + (t * 16 + l15) * 32;
            v16bf bb = load_b(bl, 0, half);
            acc[0][t] = wmma_bf16(a0, bb, acc[0][t]);
            acc[1][t] = wmma_bf16(a1, bb, acc[1][t]);
        }
    }

    const bool isV = (n0 >= 2048);
#pragma unroll
    for (int r = 0; r < 2; ++r)
#pragma unroll
        for (int t = 0; t < 4; ++t) {
            const int col = n0 + t * 16 + l15;
#pragma unroll
            for (int i = 0; i < 8; ++i) {
                const int g = m0 + 16 * r + i + 8 * half;     // global row in [0,4096)
                const unsigned short v = f2bf(acc[r][t][i]);
                if (!isV) {
                    kqv[(size_t)g * 3072 + col] = v;
                } else {
                    const int b = g >> 11, s = g & (S_LEN - 1);
                    vt[((size_t)(b * 1024 + (col - 2048))) * S_LEN + s] = v;
                }
            }
        }
}

// ---------------- Flash attention with ALiBi: one 16-query tile per wave, 64-key blocks ----------------
__global__ __launch_bounds__(256) void attn_kernel(const unsigned short* __restrict__ kqv,
                                                   const unsigned short* __restrict__ vt,
                                                   float* __restrict__ out) {
    const int lane = threadIdx.x & 31;
    const int half = lane >> 4;
    const int l15  = lane & 15;
    const int wave = threadIdx.x >> 5;
    const int gw   = blockIdx.x * 8 + wave;   // 4096 waves
    const int qt   = gw & 127;
    const int bh   = gw >> 7;
    const int b    = bh >> 4;
    const int h    = bh & 15;
    const int q0   = qt * 16;

    __shared__ __align__(16) unsigned short plds[8][16 * 64];
    unsigned short* myp = plds[wave];

    const unsigned short* qbase = kqv + (size_t)b * S_LEN * 3072 + 1024 + h * HDIM;
    const unsigned short* kbase = kqv + (size_t)b * S_LEN * 3072 + h * HDIM;
    const unsigned short* vbase = vt  + (size_t)((b * NHEAD + h) * HDIM) * S_LEN;

    const unsigned short* qrow = qbase + (size_t)(q0 + l15) * 3072;
    const v16bf qa0 = load_a(qrow, 0, half);
    const v16bf qa1 = load_a(qrow, 32, half);

    v8f o[4];
    float m8[8], l8[8];
#pragma unroll
    for (int i = 0; i < 8; ++i) {
        o[0][i] = 0.f; o[1][i] = 0.f; o[2][i] = 0.f; o[3][i] = 0.f;
        m8[i] = -__builtin_inff();
        l8[i] = 0.f;
    }

    // exp2 domain: score2 = qk * c1 + c2*(j-i);  c1 = log2e/sqrt(1024), c2 = log2e * 2^(-(h+1)/2)
    const float c1 = 0.03125f * LOG2E;
    const float c2 = LOG2E * exp2f(-0.5f * (float)(h + 1));
    const int jmax = q0 + 15;

    for (int j0 = 0; j0 <= jmax; j0 += 64) {
        // ---- scores: 16q x 64k, contraction over D=64 ----
        v8f s[4];
#pragma unroll
        for (int t = 0; t < 4; ++t)
#pragma unroll
            for (int i = 0; i < 8; ++i) s[t][i] = 0.f;
#pragma unroll
        for (int t = 0; t < 4; ++t) {
            const unsigned short* kr = kbase + (size_t)(j0 + t * 16 + l15) * 3072;
            s[t] = wmma_bf16(qa0, load_b(kr, 0, half), s[t]);
            s[t] = wmma_bf16(qa1, load_b(kr, 32, half), s[t]);
        }

        // ---- online softmax (row = i + 8*half, 16 lanes per row) ----
#pragma unroll
        for (int i = 0; i < 8; ++i) {
            const int irow = q0 + i + 8 * half;
            float sc[4];
#pragma unroll
            for (int t = 0; t < 4; ++t) {
                const int jg = j0 + t * 16 + l15;
                sc[t] = s[t][i] * c1 + c2 * (float)(jg - irow);
                if (jg > irow) sc[t] = -__builtin_inff();
            }
            float mx = fmaxf(fmaxf(sc[0], sc[1]), fmaxf(sc[2], sc[3]));
            mx = fmaxf(mx, __shfl_xor(mx, 1, 32));
            mx = fmaxf(mx, __shfl_xor(mx, 2, 32));
            mx = fmaxf(mx, __shfl_xor(mx, 4, 32));
            mx = fmaxf(mx, __shfl_xor(mx, 8, 32));
            const float mnew = fmaxf(m8[i], mx);
            const float corr = exp2f(m8[i] - mnew);
            float p[4], rs = 0.f;
            const int r = i + 8 * half;
#pragma unroll
            for (int t = 0; t < 4; ++t) {
                p[t] = exp2f(sc[t] - mnew);
                rs += p[t];
                myp[r * 64 + t * 16 + l15] = f2bf(p[t]);
            }
            rs += __shfl_xor(rs, 1, 32);
            rs += __shfl_xor(rs, 2, 32);
            rs += __shfl_xor(rs, 4, 32);
            rs += __shfl_xor(rs, 8, 32);
            l8[i] = l8[i] * corr + rs;
            m8[i] = mnew;
            o[0][i] *= corr; o[1][i] *= corr; o[2][i] *= corr; o[3][i] *= corr;
        }

        // ---- P: C-layout -> A-layout via wave-private LDS (DS ops in-order per wave) ----
        const unsigned short* prow = myp + l15 * 64;
        const v16bf pa0 = load_a(prow, 0, half);    // keys j0+0..31
        const v16bf pa1 = load_a(prow, 32, half);   // keys j0+32..63

        // ---- O += P * V (V d-major, contiguous along keys) ----
#pragma unroll
        for (int t = 0; t < 4; ++t) {
            const unsigned short* vr = vbase + (size_t)(t * 16 + l15) * S_LEN;
            o[t] = wmma_bf16(pa0, load_b(vr, j0, half), o[t]);
            o[t] = wmma_bf16(pa1, load_b(vr, j0 + 32, half), o[t]);
        }
    }

    // ---- finalize: out[b, s, h*64+d] = O / l ----
    float* obase = out + (size_t)(b * S_LEN) * EMBD + h * HDIM;
#pragma unroll
    for (int i = 0; i < 8; ++i) {
        const int row = q0 + i + 8 * half;
        const float inv = 1.0f / l8[i];
        float* orow = obase + (size_t)row * EMBD;
#pragma unroll
        for (int t = 0; t < 4; ++t) orow[t * 16 + l15] = o[t][i] * inv;
    }
}

extern "C" void kernel_launch(void* const* d_in, const int* in_sizes, int n_in,
                              void* d_out, int out_size, void* d_ws, size_t ws_size,
                              hipStream_t stream) {
    const float* x = (const float*)d_in[0];   // [2, 2048, 1024]
    const float* w = (const float*)d_in[1];   // [3072, 1024]
    float* out = (float*)d_out;               // [2, 2048, 1024]

    char* ws = (char*)d_ws;
    const size_t NX = (size_t)BSZ * S_LEN * EMBD;      // 4194304
    const size_t NW = (size_t)3 * EMBD * EMBD;         // 3145728
    unsigned short* xbf = (unsigned short*)(ws);
    unsigned short* wbf = (unsigned short*)(ws + NX * 2);
    unsigned short* kqv = (unsigned short*)(ws + NX * 2 + NW * 2);
    unsigned short* vtb = (unsigned short*)(ws + NX * 2 + NW * 2 + (size_t)BSZ * S_LEN * 3072 * 2);

    cvt_bf16_kernel<<<(int)((NX + 255) / 256), 256, 0, stream>>>(x, xbf, (int)NX);
    cvt_bf16_kernel<<<(int)((NW + 255) / 256), 256, 0, stream>>>(w, wbf, (int)NW);

    // 16 M-blocks * 48 N-tiles = 768 blocks, 256 rows x 64 cols each
    qkv_gemm_kernel<<<768, 256, 0, stream>>>(xbf, wbf, kqv, vtb);

    // 2*16 (b,h) * 128 q-tiles = 4096 waves, 8 waves/block
    attn_kernel<<<4096 / 8, 256, 0, stream>>>(kqv, vtb, out);
}
